// MultiHeadAttentionLayer_11871289606156
// MI455X (gfx1250) — compile-verified, matrix operand order checked by reading, not run on hardware
//
#include <hip/hip_runtime.h>
#include <math.h>

#ifndef __has_builtin
#define __has_builtin(x) 0
#endif

// Problem constants (from reference): B=8, S=1024, HID=768, NH=12, HD=64
constexpr int Bb  = 8;
constexpr int Ss  = 1024;
constexpr int HID = 768;
constexpr int NH  = 12;
constexpr int HD  = 64;

constexpr int BK  = 32;   // K-step per WMMA (bf16 16x16x32)
constexpr int SKP = 40;   // padded LDS row stride in ushorts (32 + 8 pad)

typedef __attribute__((ext_vector_type(16))) __bf16 v16bf;
typedef __attribute__((ext_vector_type(8)))  float  v8f;

struct U32x8 { uint4 a, b; };

__device__ __forceinline__ unsigned short f2bf(float f) {
  union { float f; unsigned u; } v; v.f = f;
  unsigned r = v.u + 0x7FFFu + ((v.u >> 16) & 1u);   // round-to-nearest-even
  return (unsigned short)(r >> 16);
}

// Load one 16x16x32 bf16 WMMA operand fragment from LDS.
// Per documented layout: lane L -> row/col L%16, K-chunks [8h,8h+8) and
// [16+8h,16+8h+8) with h=L/16 -> two 16-byte ds loads.
__device__ __forceinline__ v16bf ld_frag(const unsigned short* p) {
  U32x8 u;
  u.a = *(const uint4*)p;
  u.b = *(const uint4*)(p + 16);
  return __builtin_bit_cast(v16bf, u);
}

// ---- CDNA5 async global->LDS copy (16 bytes per lane, ASYNCcnt) ----------
__device__ __forceinline__ void async_cp16(void* lds, const void* g) {
  const unsigned l = (unsigned)(unsigned long long)lds;
  asm volatile("global_load_async_to_lds_b128 %0, %1, off"
               :: "v"(l), "v"(g) : "memory");
}
__device__ __forceinline__ void async_wait0() {
#if __has_builtin(__builtin_amdgcn_s_wait_asynccnt)
  __builtin_amdgcn_s_wait_asynccnt(0);
#else
  asm volatile("s_wait_asynccnt 0x0" ::: "memory");
#endif
}

// ---- CDNA5 Tensor Data Mover: 2D bf16 tile -> padded LDS -----------------
// D# per CDNA5 ISA. pad_i/pad_a are the encoded codes:
// pad every 2^(pad_i+1) DWORDs by (pad_a+1) DWORDs.
#if __has_builtin(__builtin_amdgcn_tensor_load_to_lds)
#define HAVE_TDM 1
typedef unsigned int u32x4 __attribute__((ext_vector_type(4)));
typedef int i32x4 __attribute__((ext_vector_type(4)));
typedef int i32x8 __attribute__((ext_vector_type(8)));

__device__ __forceinline__ void tdm_load_tile_bf16(
    void* lds, const void* gaddr, int rows, int rowlen, int stride_elems,
    int pad_i, int pad_a)
{
  const unsigned long long ga = (unsigned long long)gaddr;
  u32x4 g0;
  g0[0] = 1u;                                            // count=1 (valid D#)
  g0[1] = (unsigned)(unsigned long long)lds;             // lds_addr (bytes)
  g0[2] = (unsigned)(ga & 0xFFFFFFFFu);                  // global_addr[31:0]
  g0[3] = (unsigned)((ga >> 32) & 0x1FFFFFFu) | (2u << 30); // ga[56:32], type=2
  i32x8 g1;
  g1[0] = (int)(0x10000u |                // data_size = 1 (2 bytes)
                0x100000u |               // pad_enable
                ((unsigned)pad_i << 22) | ((unsigned)pad_a << 25));
  g1[1] = (int)((unsigned)rowlen << 16);  // tensor_dim0
  g1[2] = (int)((unsigned)rows << 16);    // tensor_dim1
  g1[3] = (int)((unsigned)rowlen << 16);  // tile_dim0
  g1[4] = rows;                           // tile_dim1 (tile_dim2 = 0)
  g1[5] = stride_elems;                   // tensor_dim0_stride
  g1[6] = 0;
  g1[7] = 0;
  const i32x4 gz = {0, 0, 0, 0};
#if __clang_major__ >= 23
  const i32x8 gz8 = {0, 0, 0, 0, 0, 0, 0, 0};
  __builtin_amdgcn_tensor_load_to_lds(g0, g1, gz, gz, gz8, 0);
#else
  __builtin_amdgcn_tensor_load_to_lds(g0, g1, gz, gz, 0);
#endif
}
#endif

// ---------------------------------------------------------------------------
// Generic WMMA GEMM for the dense projections / FFN:
//   C = A(f32, MxK row-major) @ W(f32, KxN row-major) + bias
//   OT = float or ushort (bf16 raw)
// TOUT: write output transposed as [B][N][Ss] (used for V^T so the attention
// kernel can stage V tiles with plain strided TDM copies, no transpose).
// Requires M % 128 == 0, N % 128 == 0, K % 32 == 0 (true for all calls).
// ---------------------------------------------------------------------------
template <typename OT, bool TOUT = false>
__global__ __launch_bounds__(256) void wmma_gemm(
    const float* __restrict__ A, const float* __restrict__ W,
    const float* __restrict__ bias, OT* __restrict__ C,
    int K, int lda, int ldw, int ldc)
{
  constexpr int BM = 128, BN = 128;
  __shared__ unsigned short As[BM * SKP];
  __shared__ unsigned short Bs[BN * SKP];

  const int tid  = threadIdx.x;
  const int lane = tid & 31;
  const int wid  = tid >> 5;
  const int wm   = wid & 3;          // 4 waves tile M
  const int wn   = wid >> 2;         // 2 waves tile N
  constexpr int WM = BM / 4;         // 32
  constexpr int WN = BN / 2;         // 64
  constexpr int FM = WM / 16;        // 2
  constexpr int FN = WN / 16;        // 4

  const int blockM = blockIdx.y * BM;
  const int blockN = blockIdx.x * BN;
  const int m_h = lane >> 4;
  const int m_r = lane & 15;

  v8f acc[FM][FN];
  #pragma unroll
  for (int fm = 0; fm < FM; fm++)
    #pragma unroll
    for (int fn = 0; fn < FN; fn++)
      #pragma unroll
      for (int i = 0; i < 8; i++) acc[fm][fn][i] = 0.0f;

  for (int k0 = 0; k0 < K; k0 += BK) {
    // stage A: 128x32, f32 -> bf16
    #pragma unroll
    for (int i = tid; i < BM * BK / 4; i += 256) {
      const int row = i >> 3, q = i & 7;
      const float* src = A + (long)(blockM + row) * lda + k0 + q * 4;
      float4 f = *(const float4*)src;
      if (k0 + BK < K) __builtin_prefetch(src + BK, 0, 0);
      unsigned short* dst = &As[row * SKP + q * 4];
      dst[0] = f2bf(f.x); dst[1] = f2bf(f.y); dst[2] = f2bf(f.z); dst[3] = f2bf(f.w);
    }
    // stage B (KxN row-major) transposed into Bs[n][k]
    #pragma unroll
    for (int i = tid; i < BK * BN; i += 256) {
      const int k = i / BN, n = i % BN;
      Bs[n * SKP + k] = f2bf(W[(long)(k0 + k) * ldw + blockN + n]);
    }
    __syncthreads();

    v16bf afr[FM];
    #pragma unroll
    for (int fm = 0; fm < FM; fm++)
      afr[fm] = ld_frag(&As[(wm * WM + fm * 16 + m_r) * SKP + m_h * 8]);
    v16bf bfr[FN];
    #pragma unroll
    for (int fn = 0; fn < FN; fn++)
      bfr[fn] = ld_frag(&Bs[(wn * WN + fn * 16 + m_r) * SKP + m_h * 8]);

    #pragma unroll
    for (int fm = 0; fm < FM; fm++)
      #pragma unroll
      for (int fn = 0; fn < FN; fn++)
        acc[fm][fn] = __builtin_amdgcn_wmma_f32_16x16x32_bf16(
            false, afr[fm], false, bfr[fn], (short)0, acc[fm][fn], false, false);

    __syncthreads();
  }

  #pragma unroll
  for (int fm = 0; fm < FM; fm++)
    #pragma unroll
    for (int fn = 0; fn < FN; fn++) {
      const int nG = blockN + wn * WN + fn * 16 + m_r;
      const float bv = bias[nG];
      #pragma unroll
      for (int r = 0; r < 8; r++) {
        const int mG = blockM + wm * WM + fm * 16 + r + 8 * m_h;
        const float val = acc[fm][fn][r] + bv;
        if constexpr (TOUT) {
          // mG = (b, s); store C[b][nG][s], b = mG/Ss, s = mG%Ss
          C[((long)(mG >> 10) * HID + nG) * Ss + (mG & (Ss - 1))] = (OT)f2bf(val);
        } else if constexpr (sizeof(OT) == 2) {
          C[(long)mG * ldc + nG] = (OT)f2bf(val);
        } else {
          C[(long)mG * ldc + nG] = (OT)val;
        }
      }
    }
}

// ---------------------------------------------------------------------------
// Fused attention: per block = one (b, h, 64-query-row strip).
//   E = (Q K^T)/8 with mask  -> kept entirely in VGPRs (64x1024 f32 strip)
//   softmax (max, exp, sum) in registers with LDS cross-wave reductions
//   P written to d_out exactly once (non-temporal); P tile re-fragmented via
//   LDS; PV accumulated with WMMA in the same pass.
// Q strip staged with async global->LDS b128 copies (ASYNCcnt).
// K and V^T tiles staged with the Tensor Data Mover, DOUBLE-BUFFERED:
//   issue tile t+1, s_wait_tensorcnt 1 (t done, t+1 in flight during compute).
// 512 threads = 16 waves. Q/K/V^T are bf16 (L2-resident), mask int32.
// ---------------------------------------------------------------------------
__global__ __launch_bounds__(512) void attn_fused(
    const unsigned short* __restrict__ Qbf,
    const unsigned short* __restrict__ Kbf,
    const unsigned short* __restrict__ VbfT,  // [B][HID][Ss] bf16
    const int* __restrict__ mask,
    float* __restrict__ attnP,    // [B,NH,S,S]
    float* __restrict__ AVo)      // [B,S,HID]
{
  constexpr int QSK = 72;    // 64 + 8 pad (ushorts)  == TDM pad 4dw/32dw
  constexpr int PSK = 136;   // 128 + 8 pad (ushorts) == TDM pad 4dw/64dw

  __shared__ unsigned short Qs[64 * QSK];
  __shared__ unsigned short Ks[2][128 * QSK];
  __shared__ unsigned short Pt[64 * PSK];
  __shared__ unsigned short Vs[2][64 * PSK];
  __shared__ float part[64 * 8];
  __shared__ float rowm[64];
  __shared__ float rowsum[64];

  const int strip = blockIdx.x;            // 0..15
  const int zz    = blockIdx.y;            // 0..B*NH-1
  const int b = zz / NH, h = zz % NH;
  const int row0 = strip * 64;

  const unsigned short* Qp  = Qbf + (long)b * Ss * HID + h * HD;
  const unsigned short* Kp  = Kbf + (long)b * Ss * HID + h * HD;
  const unsigned short* Vtp = VbfT + ((long)b * HID + h * HD) * Ss;
  const int* mp = mask + (long)b * Ss;
  float* Pout = attnP + ((long)zz * Ss + row0) * Ss;
  float* Op   = AVo + ((long)b * Ss + row0) * HID + h * HD;

  const int tid  = threadIdx.x;
  const int lane = tid & 31;
  const int w    = tid >> 5;
  const int wm = w & 1, wn = w >> 1;       // E mapping
  const int w4 = w & 3, wd = w >> 2;       // PV mapping
  const int m_h = lane >> 4, m_r = lane & 15;

  // ---- stage Q strip (64x64 bf16) via async global->LDS b128 ----
  #pragma unroll
  for (int i = tid; i < 64 * 8; i += 512) {     // 8 x 16B chunks per row
    const int row = i >> 3, q = i & 7;
    async_cp16(&Qs[row * QSK + q * 8], Qp + (long)(row0 + row) * HID + q * 8);
  }
#ifdef HAVE_TDM
  if (w == 0)   // prologue: start DMA of K tile 0 while Q copy is in flight
    tdm_load_tile_bf16(Ks[0], Kp, 128, 64, HID, 4, 3);
#endif

  // ---- init E while the copies are in flight ----
  v8f E[8][2];
  #pragma unroll
  for (int t = 0; t < 8; t++)
    #pragma unroll
    for (int fm = 0; fm < 2; fm++)
      #pragma unroll
      for (int i = 0; i < 8; i++) E[t][fm][i] = 0.0f;

  async_wait0();
  __syncthreads();
  v16bf afrQ[2][2];
  #pragma unroll
  for (int fm = 0; fm < 2; fm++)
    #pragma unroll
    for (int ks = 0; ks < 2; ks++)
      afrQ[fm][ks] = ld_frag(&Qs[(wm * 32 + fm * 16 + m_r) * QSK + ks * 32 + m_h * 8]);

  // ---- sweep 1: E = (Q K^T)/8 + mask, resident in VGPRs ----
  #pragma unroll
  for (int t = 0; t < 8; t++) {
    __syncthreads();   // readers of the buffer tile t+1 will overwrite are done
#ifdef HAVE_TDM
    if (w == 0) {
      if (t < 7) {
        tdm_load_tile_bf16(Ks[(t + 1) & 1], Kp + (long)(t + 1) * 128 * HID,
                           128, 64, HID, 4, 3);
        __builtin_amdgcn_s_wait_tensorcnt(1);   // tile t done; t+1 in flight
      } else {
        __builtin_amdgcn_s_wait_tensorcnt(0);
      }
    }
#else
    #pragma unroll
    for (int i = tid; i < 128 * 8; i += 512) {
      const int row = i >> 3, q = i & 7;
      async_cp16(&Ks[t & 1][row * QSK + q * 8],
                 Kp + (long)(t * 128 + row) * HID + q * 8);
    }
    async_wait0();
#endif
    __syncthreads();   // tile t visible to all waves
    #pragma unroll
    for (int ks = 0; ks < 2; ks++) {
      v16bf bfr = ld_frag(&Ks[t & 1][(wn * 16 + m_r) * QSK + ks * 32 + m_h * 8]);
      #pragma unroll
      for (int fm = 0; fm < 2; fm++)
        E[t][fm] = __builtin_amdgcn_wmma_f32_16x16x32_bf16(
            false, afrQ[fm][ks], false, bfr, (short)0, E[t][fm], false, false);
    }
    const int mv = mp[t * 128 + wn * 16 + m_r];
    #pragma unroll
    for (int fm = 0; fm < 2; fm++)
      #pragma unroll
      for (int i = 0; i < 8; i++) {
        const float e = E[t][fm][i] * 0.125f;
        E[t][fm][i] = (mv == 0) ? -1.0e10f : e;
      }
  }

  // ---- row max (16-lane shuffle reduce + LDS partials) ----
  #pragma unroll
  for (int fm = 0; fm < 2; fm++)
    #pragma unroll
    for (int r = 0; r < 8; r++) {
      float mx = -3.0e38f;
      #pragma unroll
      for (int t = 0; t < 8; t++) mx = fmaxf(mx, E[t][fm][r]);
      for (int o = 8; o > 0; o >>= 1) mx = fmaxf(mx, __shfl_xor(mx, o, 32));
      if (m_r == 0) part[(wm * 32 + fm * 16 + r + 8 * m_h) * 8 + wn] = mx;
    }
  __syncthreads();
  if (tid < 64) {
    float m = part[tid * 8];
    #pragma unroll
    for (int j = 1; j < 8; j++) m = fmaxf(m, part[tid * 8 + j]);
    rowm[tid] = m;
  }
  __syncthreads();

  // ---- sweep 2: exp + row sum, pure register math ----
  #pragma unroll
  for (int fm = 0; fm < 2; fm++)
    #pragma unroll
    for (int r = 0; r < 8; r++) {
      const float m = rowm[wm * 32 + fm * 16 + r + 8 * m_h];
      float s = 0.0f;
      #pragma unroll
      for (int t = 0; t < 8; t++) {
        const float p = __expf(E[t][fm][r] - m);
        E[t][fm][r] = p;
        s += p;
      }
      for (int o = 8; o > 0; o >>= 1) s += __shfl_xor(s, o, 32);
      if (m_r == 0) part[(wm * 32 + fm * 16 + r + 8 * m_h) * 8 + wn] = s;
    }
  __syncthreads();
  if (tid < 64) {
    float s = part[tid * 8];
    #pragma unroll
    for (int j = 1; j < 8; j++) s += part[tid * 8 + j];
    rowsum[tid] = s;
  }
  __syncthreads();

  float invr[2][8];
  #pragma unroll
  for (int fm = 0; fm < 2; fm++)
    #pragma unroll
    for (int r = 0; r < 8; r++)
      invr[fm][r] = 1.0f / rowsum[wm * 32 + fm * 16 + r + 8 * m_h];

  // ---- sweep 3: write P once, re-fragment via LDS, accumulate PV ----
  v8f Oacc;
  #pragma unroll
  for (int i = 0; i < 8; i++) Oacc[i] = 0.0f;

#ifdef HAVE_TDM
  if (w == 0)   // prologue: start DMA of V^T tile 0 (64 rows x 128, stride Ss)
    tdm_load_tile_bf16(Vs[0], Vtp, 64, 128, Ss, 5, 3);
#endif

  #pragma unroll
  for (int t = 0; t < 8; t++) {
    __syncthreads();   // PV readers of Pt / other V buffer are done
#ifdef HAVE_TDM
    if (w == 0 && t < 7)
      tdm_load_tile_bf16(Vs[(t + 1) & 1], Vtp + (long)(t + 1) * 128,
                         64, 128, Ss, 5, 3);
#else
    #pragma unroll
    for (int i = tid; i < 64 * 16; i += 512) {   // 16 x 16B chunks per row
      const int d = i >> 4, q = i & 15;
      async_cp16(&Vs[t & 1][d * PSK + q * 8],
                 Vtp + (long)d * Ss + t * 128 + q * 8);
    }
#endif
    // normalized P: registers -> global (non-temporal, write-once stream)
    // + bf16 tile in LDS; overlaps the V DMA.
    #pragma unroll
    for (int fm = 0; fm < 2; fm++)
      #pragma unroll
      for (int r = 0; r < 8; r++) {
        const int rl = wm * 32 + fm * 16 + r + 8 * m_h;
        const int cl = wn * 16 + m_r;
        const float p = E[t][fm][r] * invr[fm][r];
        Pt[rl * PSK + cl] = f2bf(p);
        __builtin_nontemporal_store(p, &Pout[(long)rl * Ss + t * 128 + cl]);
      }
#ifdef HAVE_TDM
    if (w == 0) {
      if (t < 7) __builtin_amdgcn_s_wait_tensorcnt(1);
      else       __builtin_amdgcn_s_wait_tensorcnt(0);
    }
#else
    async_wait0();
#endif
    __syncthreads();   // V tile t + P tile visible
    #pragma unroll
    for (int kc = 0; kc < 4; kc++) {
      v16bf afr = ld_frag(&Pt[(w4 * 16 + m_r) * PSK + kc * 32 + m_h * 8]);
      v16bf bfr = ld_frag(&Vs[t & 1][(wd * 16 + m_r) * PSK + kc * 32 + m_h * 8]);
      Oacc = __builtin_amdgcn_wmma_f32_16x16x32_bf16(
          false, afr, false, bfr, (short)0, Oacc, false, false);
    }
  }

  #pragma unroll
  for (int r = 0; r < 8; r++)
    Op[(long)(w4 * 16 + r + 8 * m_h) * HID + wd * 16 + m_r] = Oacc[r];
}

// xcat[:, 768:1536] = query
__global__ __launch_bounds__(256) void concat_copy_kernel(
    const float* __restrict__ q, float* __restrict__ xcat, int total)
{
  const int i = blockIdx.x * 256 + threadIdx.x;
  if (i < total) {
    const int row = i / HID, c = i % HID;
    xcat[(long)row * (2 * HID) + HID + c] = q[i];
  }
}

// out = sigmoid(h1) * h2   (write-once stream -> non-temporal)
__global__ __launch_bounds__(256) void glu_out_kernel(
    const float* __restrict__ h1, const float* __restrict__ h2,
    float* __restrict__ out, int total)
{
  const int i = blockIdx.x * 256 + threadIdx.x;
  if (i < total) {
    const float a = h1[i];
    __builtin_nontemporal_store((1.0f / (1.0f + __expf(-a))) * h2[i], &out[i]);
  }
}

extern "C" void kernel_launch(void* const* d_in, const int* in_sizes, int n_in,
                              void* d_out, int out_size, void* d_ws, size_t ws_size,
                              hipStream_t stream) {
  (void)in_sizes; (void)n_in; (void)out_size; (void)ws_size;
  const float* query = (const float*)d_in[0];
  const float* key   = (const float*)d_in[1];
  const float* value = (const float*)d_in[2];
  const int*   mask  = (const int*)d_in[3];
  const float* Wq = (const float*)d_in[4];  const float* bq = (const float*)d_in[5];
  const float* Wk = (const float*)d_in[6];  const float* bk = (const float*)d_in[7];
  const float* Wv = (const float*)d_in[8];  const float* bv = (const float*)d_in[9];
  const float* Wo = (const float*)d_in[10]; const float* bo = (const float*)d_in[11];
  const float* W1 = (const float*)d_in[12]; const float* b1 = (const float*)d_in[13];
  const float* W2 = (const float*)d_in[14]; const float* b2 = (const float*)d_in[15];

  float* out  = (float*)d_out;                       // [B,S,HID]
  float* attn = out + (long)Bb * Ss * HID;           // [B,NH,S,S]

  const long MS = (long)Bb * Ss;                     // 8192 rows
  char* ws = (char*)d_ws;
  unsigned short* Qbf  = (unsigned short*)ws;        // bf16 (L2-resident)
  unsigned short* Kbf  = Qbf + MS * HID;
  unsigned short* VbfT = Kbf + MS * HID;             // [B][HID][Ss] bf16
  float* AVo  = (float*)(ws + 3 * MS * HID * sizeof(unsigned short)); // attn@V f32
  float* xcat = AVo + MS * HID;                      // f32 [B,S,1536]
  float* H1   = AVo;                                 // reuse (AVo dead after Wo gemm)
  float* H2   = (float*)ws;                          // reuse Q/K region (dead after attn)

  dim3 blk(256, 1, 1);

  // 1) Q/K/V projections -> bf16 (V written transposed for TDM staging)
  dim3 gp(HID / 128, (int)(MS / 128), 1);
  wmma_gemm<unsigned short><<<gp, blk, 0, stream>>>(query, Wq, bq, Qbf, HID, HID, HID, HID);
  wmma_gemm<unsigned short><<<gp, blk, 0, stream>>>(key,   Wk, bk, Kbf, HID, HID, HID, HID);
  wmma_gemm<unsigned short, true><<<gp, blk, 0, stream>>>(value, Wv, bv, VbfT, HID, HID, HID, HID);

  // 2) fused energy + mask + softmax + P-write + PV
  dim3 gf(Ss / 64, Bb * NH, 1);
  attn_fused<<<gf, dim3(512, 1, 1), 0, stream>>>(Qbf, Kbf, VbfT, mask, attn, AVo);

  // 3) xcat[:, :768] = AVo @ Wo + bo
  wmma_gemm<float><<<gp, blk, 0, stream>>>(AVo, Wo, bo, xcat, HID, HID, HID, 2 * HID);

  // 4) xcat[:, 768:] = query
  const int total = (int)(MS * HID);
  concat_copy_kernel<<<dim3((total + 255) / 256, 1, 1), blk, 0, stream>>>(query, xcat, total);

  // 5) FFN: H1 = xcat @ W1 + b1 ; H2 = xcat @ W2 + b2
  wmma_gemm<float><<<gp, blk, 0, stream>>>(xcat, W1, b1, H1, 2 * HID, 2 * HID, HID, HID);
  wmma_gemm<float><<<gp, blk, 0, stream>>>(xcat, W2, b2, H2, 2 * HID, 2 * HID, HID, HID);

  // 6) out = sigmoid(H1) * H2
  glu_out_kernel<<<dim3((total + 255) / 256, 1, 1), blk, 0, stream>>>(H1, H2, out, total);
}
